// Vimfeat_23476291240107
// MI455X (gfx1250) — matrix operand order repaired and checked by reading, not compiled, not run on hardware
//
#include <hip/hip_runtime.h>
#include <hip/hip_bf16.h>

// ---------------------------------------------------------------------------
// Model constants
// ---------------------------------------------------------------------------
#define BATCH    16
#define LSEQ     495          // 494 + cls token
#define LIN      494
#define TP       247          // insertion point of cls token
#define DMODEL   384
#define DINNER   768
#define DSTATE   16
#define DTRANK   24
#define KCONV    4
#define NDEPTH   4
#define MROWS    (BATCH * LSEQ)          // 7920, divisible by 16
#define FEATN    49
#define NJOINT   14
#define NVERT    431                     // 495 - 1 - 49 - 14
#define DBLLD    64                      // padded row stride of dbl buffer

typedef __attribute__((ext_vector_type(16))) __bf16 v16bf;
typedef __attribute__((ext_vector_type(8)))  __bf16 v8bf;
typedef __attribute__((ext_vector_type(8)))  float  v8f;

// fp32 -> bf16 round-to-nearest-even (bit trick)
static __device__ __forceinline__ __bf16 tobf(float x) {
    union { float f; unsigned u; } v; v.f = x;
    unsigned r = v.u + 0x7FFFu + ((v.u >> 16) & 1u);
    union { unsigned short s; __bf16 b; } o;
    o.s = (unsigned short)(r >> 16);
    return o.b;
}

static __device__ __forceinline__ float fsilu(float x) {
    return x / (1.0f + __expf(-x));
}

// ---------------------------------------------------------------------------
// Generic fp32 -> bf16 staging with zero padding.
// dst is rows_d x cols_d (row-major, contiguous); src row stride = sld.
// Elements outside (rows_v x cols_v) are written as 0.
// ---------------------------------------------------------------------------
__global__ void cvt_bf16_kernel(const float* __restrict__ src, int sld,
                                __bf16* __restrict__ dst,
                                int rows_d, int cols_d, int rows_v, int cols_v) {
    int idx = blockIdx.x * blockDim.x + threadIdx.x;
    if (idx >= rows_d * cols_d) return;
    int c = idx % cols_d;
    int r = idx / cols_d;
    float v = (r < rows_v && c < cols_v) ? src[(size_t)r * sld + c] : 0.0f;
    dst[idx] = tobf(v);
}

// ---------------------------------------------------------------------------
// GEMM: Out[m,n] = sum_k X[m,k] * W[n,k]   (bf16 inputs, fp32 accumulate)
// One wave per block -> 16x64 output tile, 4 accumulators, A reused 4x.
// All dims are exact multiples (M%16, N%64, K%32 == 0): NO guards, pure
// vector loads + v_wmma_f32_16x16x32_bf16.
// mode 0: plain   mode 1: softplus(acc + bias[n])   mode 2: acc * 0.5
// ---------------------------------------------------------------------------
__global__ void gemm_wmma_kernel(const __bf16* __restrict__ X, int ldx,
                                 const __bf16* __restrict__ W, int ldw,
                                 float* __restrict__ Out, int ldo,
                                 int Kd, int mode,
                                 const float* __restrict__ bias) {
    const int lane = threadIdx.x;       // 0..31
    const int half = lane >> 4;
    const int lm   = lane & 15;
    const int m0   = blockIdx.x * 16;
    const int n0   = blockIdx.y * 64;

    v8f c[4];
#pragma unroll
    for (int t = 0; t < 4; ++t) c[t] = (v8f){0.f,0.f,0.f,0.f,0.f,0.f,0.f,0.f};

    // A: lanes 0-15 -> M=lane,   K = k0 + 0..15
    //    lanes16-31 -> M=lane-16,K = k0 + 16..31
    const __bf16* xr = X + (size_t)(m0 + lm) * ldx + half * 16;
    // B: lanes 0-15 -> N=lane,   K = k0 + 0..15
    //    lanes16-31 -> N=lane-16,K = k0 + 16..31
    const __bf16* wr0 = W + (size_t)(n0 + 0 * 16 + lm) * ldw + half * 16;
    const __bf16* wr1 = W + (size_t)(n0 + 1 * 16 + lm) * ldw + half * 16;
    const __bf16* wr2 = W + (size_t)(n0 + 2 * 16 + lm) * ldw + half * 16;
    const __bf16* wr3 = W + (size_t)(n0 + 3 * 16 + lm) * ldw + half * 16;

    for (int k0 = 0; k0 < Kd; k0 += 32) {
        v8bf alo = *(const v8bf*)(xr + k0);
        v8bf ahi = *(const v8bf*)(xr + k0 + 8);
        v16bf a = __builtin_shufflevector(alo, ahi,
                      0,1,2,3,4,5,6,7,8,9,10,11,12,13,14,15);

        v8bf b0lo = *(const v8bf*)(wr0 + k0);
        v8bf b0hi = *(const v8bf*)(wr0 + k0 + 8);
        v16bf b0 = __builtin_shufflevector(b0lo, b0hi,
                      0,1,2,3,4,5,6,7,8,9,10,11,12,13,14,15);
        c[0] = __builtin_amdgcn_wmma_f32_16x16x32_bf16(
                   false, a, false, b0, (short)0, c[0], false, false);

        v8bf b1lo = *(const v8bf*)(wr1 + k0);
        v8bf b1hi = *(const v8bf*)(wr1 + k0 + 8);
        v16bf b1 = __builtin_shufflevector(b1lo, b1hi,
                      0,1,2,3,4,5,6,7,8,9,10,11,12,13,14,15);
        c[1] = __builtin_amdgcn_wmma_f32_16x16x32_bf16(
                   false, a, false, b1, (short)0, c[1], false, false);

        v8bf b2lo = *(const v8bf*)(wr2 + k0);
        v8bf b2hi = *(const v8bf*)(wr2 + k0 + 8);
        v16bf b2 = __builtin_shufflevector(b2lo, b2hi,
                      0,1,2,3,4,5,6,7,8,9,10,11,12,13,14,15);
        c[2] = __builtin_amdgcn_wmma_f32_16x16x32_bf16(
                   false, a, false, b2, (short)0, c[2], false, false);

        v8bf b3lo = *(const v8bf*)(wr3 + k0);
        v8bf b3hi = *(const v8bf*)(wr3 + k0 + 8);
        v16bf b3 = __builtin_shufflevector(b3lo, b3hi,
                      0,1,2,3,4,5,6,7,8,9,10,11,12,13,14,15);
        c[3] = __builtin_amdgcn_wmma_f32_16x16x32_bf16(
                   false, a, false, b3, (short)0, c[3], false, false);
    }

    // C/D layout: VGPR j -> row m0 + j + 8*half, col n0 + t*16 + (lane&15)
#pragma unroll
    for (int t = 0; t < 4; ++t) {
        int col = n0 + t * 16 + lm;
        float bb = (mode == 1) ? bias[col] : 0.0f;
#pragma unroll
        for (int j = 0; j < 8; ++j) {
            int row = m0 + j + half * 8;
            float val = c[t][j];
            if (mode == 1) {
                float xx = val + bb;                       // softplus
                val = (xx > 20.0f) ? xx : log1pf(__expf(xx));
            } else if (mode == 2) {
                val *= 0.5f;
            }
            Out[(size_t)row * ldo + col] = val;
        }
    }
}

// ---------------------------------------------------------------------------
// Build hidden = concat(cjv[:, :TP], cls, cjv[:, TP:]) + pos_embed ; residual = 0
// ---------------------------------------------------------------------------
__global__ void build_kernel(const float* __restrict__ cjv,
                             const float* __restrict__ cls,
                             const float* __restrict__ pos,
                             float* __restrict__ hidden,
                             float* __restrict__ residual) {
    int idx = blockIdx.x * blockDim.x + threadIdx.x;
    if (idx >= MROWS * DMODEL) return;
    int d   = idx % DMODEL;
    int row = idx / DMODEL;
    int b   = row / LSEQ;
    int l   = row % LSEQ;
    float v;
    if (l < TP)        v = cjv[((size_t)b * LIN + l) * DMODEL + d];
    else if (l == TP)  v = cls[d];
    else               v = cjv[((size_t)b * LIN + (l - 1)) * DMODEL + d];
    v += pos[(size_t)l * DMODEL + d];
    hidden[idx]   = v;
    residual[idx] = 0.0f;
}

// ---------------------------------------------------------------------------
// residual += hidden ; out_bf16 = rmsnorm(residual) * w   (one block per row)
// Writes bf16 directly: the only consumer is the in_proj WMMA GEMM.
// ---------------------------------------------------------------------------
__global__ void rms_add_kernel(float* __restrict__ residual,
                               const float* __restrict__ hidden,
                               const float* __restrict__ w,
                               __bf16* __restrict__ out) {
    const int row = blockIdx.x;
    const int tid = threadIdx.x;              // 256 threads
    const size_t base = (size_t)row * DMODEL;

    float r0 = residual[base + tid] + hidden[base + tid];
    residual[base + tid] = r0;
    float r1 = 0.0f;
    if (tid < DMODEL - 256) {
        r1 = residual[base + 256 + tid] + hidden[base + 256 + tid];
        residual[base + 256 + tid] = r1;
    }
    __shared__ float red[256];
    red[tid] = r0 * r0 + r1 * r1;
    __syncthreads();
    for (int s = 128; s > 0; s >>= 1) {
        if (tid < s) red[tid] += red[tid + s];
        __syncthreads();
    }
    float scale = rsqrtf(red[0] / (float)DMODEL + 1e-5f);
    out[base + tid] = tobf(r0 * scale * w[tid]);
    if (tid < DMODEL - 256) out[base + 256 + tid] = tobf(r1 * scale * w[256 + tid]);
}

// ---------------------------------------------------------------------------
// Depthwise causal conv (K=4) + SiLU; writes fp32 (scan) and bf16 (x_proj).
// dir=1 reads the sequence reversed so downstream buffers are in scan order.
// ---------------------------------------------------------------------------
__global__ void conv_silu_kernel(const float* __restrict__ xz,
                                 const float* __restrict__ cw,
                                 const float* __restrict__ cb,
                                 float* __restrict__ xc,
                                 __bf16* __restrict__ xcbf, int dir) {
    int idx = blockIdx.x * blockDim.x + threadIdx.x;
    if (idx >= MROWS * DINNER) return;
    int d = idx % DINNER;
    int r = idx / DINNER;
    int b = r / LSEQ;
    int l = r % LSEQ;
    float s = cb[d];
#pragma unroll
    for (int k = 0; k < KCONV; ++k) {
        int j = l - (KCONV - 1) + k;
        if (j >= 0) {
            int sl = dir ? (LSEQ - 1 - j) : j;
            s += cw[d * KCONV + k] * xz[((size_t)b * LSEQ + sl) * (2 * DINNER) + d];
        }
    }
    float v = fsilu(s);
    xc[idx]   = v;
    xcbf[idx] = tobf(v);
}

// ---------------------------------------------------------------------------
// Selective scan: one block per batch element, 768 threads (one per channel),
// 16 states per thread in registers, B/C broadcast via LDS.
// Writes y at the un-reversed position, gated by silu(z) (dir=1 accumulates).
// ---------------------------------------------------------------------------
__global__ void scan_kernel(const float* __restrict__ xc,
                            const float* __restrict__ dbl,
                            const float* __restrict__ delta,
                            const float* __restrict__ xz,
                            const float* __restrict__ Alog,
                            const float* __restrict__ Dp,
                            float* __restrict__ y, int dir) {
    const int b = blockIdx.x;
    const int d = threadIdx.x;            // 0..767

    float A[DSTATE];
#pragma unroll
    for (int n = 0; n < DSTATE; ++n) A[n] = -__expf(Alog[d * DSTATE + n]);
    const float Dv = Dp[d];

    float h[DSTATE];
#pragma unroll
    for (int n = 0; n < DSTATE; ++n) h[n] = 0.0f;

    __shared__ float sB[DSTATE];
    __shared__ float sC[DSTATE];

    for (int l = 0; l < LSEQ; ++l) {
        const size_t row = (size_t)b * LSEQ + l;
        if (threadIdx.x < 2 * DSTATE) {
            int n = threadIdx.x & (DSTATE - 1);
            if (threadIdx.x < DSTATE) sB[n] = dbl[row * DBLLD + DTRANK + n];
            else                      sC[n] = dbl[row * DBLLD + DTRANK + DSTATE + n];
        }
        __syncthreads();

        float dl  = delta[row * DINNER + d];
        float xv  = xc[row * DINNER + d];
        float dbx = dl * xv;
        float acc = 0.0f;
#pragma unroll
        for (int n = 0; n < DSTATE; ++n) {
            h[n] = __expf(dl * A[n]) * h[n] + dbx * sB[n];
            acc += h[n] * sC[n];
        }

        int pos = dir ? (LSEQ - 1 - l) : l;
        size_t prow = (size_t)b * LSEQ + pos;
        float zv  = xz[prow * (2 * DINNER) + DINNER + d];
        float val = (acc + xv * Dv) * fsilu(zv);
        if (dir) y[prow * DINNER + d] += val;
        else     y[prow * DINNER + d]  = val;
        __syncthreads();
    }
}

// ---------------------------------------------------------------------------
// Final: residual += hidden ; rmsnorm(final_w) ; token reshuffle ; scatter.
// ---------------------------------------------------------------------------
__global__ void final_kernel(float* __restrict__ residual,
                             const float* __restrict__ hidden,
                             const float* __restrict__ w,
                             float* __restrict__ out) {
    const int row = blockIdx.x;
    const int tid = threadIdx.x;          // 256 threads
    const int b = row / LSEQ;
    const int l = row % LSEQ;
    const size_t base = (size_t)row * DMODEL;

    float r0 = residual[base + tid] + hidden[base + tid];
    float r1 = 0.0f;
    if (tid < DMODEL - 256)
        r1 = residual[base + 256 + tid] + hidden[base + 256 + tid];

    __shared__ float red[256];
    red[tid] = r0 * r0 + r1 * r1;
    __syncthreads();
    for (int s = 128; s > 0; s >>= 1) {
        if (tid < s) red[tid] += red[tid + s];
        __syncthreads();
    }
    float scale = rsqrtf(red[0] / (float)DMODEL + 1e-5f);

    int lr = (l == TP) ? 0 : ((l < TP) ? l + 1 : l);
    size_t obase; bool store = true;
    if (lr == 0) {
        obase = (size_t)b * DMODEL;                                   // cam
    } else if (lr < 1 + FEATN) {
        store = false;                                                // dropped
    } else if (lr < 1 + FEATN + NJOINT) {
        obase = (size_t)BATCH * DMODEL
              + ((size_t)b * NJOINT + (lr - 1 - FEATN)) * DMODEL;     // joints
    } else {
        obase = (size_t)BATCH * DMODEL + (size_t)BATCH * NJOINT * DMODEL
              + ((size_t)b * NVERT + (lr - 1 - FEATN - NJOINT)) * DMODEL; // verts
    }
    if (store) {
        out[obase + tid] = r0 * scale * w[tid];
        if (tid < DMODEL - 256) out[obase + 256 + tid] = r1 * scale * w[256 + tid];
    }
}

// ---------------------------------------------------------------------------
// Host side
// ---------------------------------------------------------------------------
static inline void launch_cvt(const float* src, int sld, __bf16* dst,
                              int rows_d, int cols_d, int rows_v, int cols_v,
                              hipStream_t stream) {
    int n = rows_d * cols_d;
    cvt_bf16_kernel<<<(n + 255) / 256, 256, 0, stream>>>(
        src, sld, dst, rows_d, cols_d, rows_v, cols_v);
}

extern "C" void kernel_launch(void* const* d_in, const int* in_sizes, int n_in,
                              void* d_out, int out_size, void* d_ws, size_t ws_size,
                              hipStream_t stream) {
    (void)in_sizes; (void)n_in; (void)out_size; (void)ws_size;

    const float* cjv  = (const float*)d_in[0];
    /* d_in[1] = feat_num (constant 49, baked in) */
    const float* cls  = (const float*)d_in[2];
    const float* pos  = (const float*)d_in[3];
    const float* inW  = (const float*)d_in[4];
    const float* cW   = (const float*)d_in[5];
    const float* cB   = (const float*)d_in[6];
    const float* xpW  = (const float*)d_in[7];
    const float* dtW  = (const float*)d_in[8];
    const float* dtB  = (const float*)d_in[9];
    const float* Alog = (const float*)d_in[10];
    const float* Dsk  = (const float*)d_in[11];
    const float* outW = (const float*)d_in[12];
    const float* nw   = (const float*)d_in[13];
    const float* fnw  = (const float*)d_in[14];
    float* out = (float*)d_out;

    float* ws = (float*)d_ws;
    const size_t MD = (size_t)MROWS * DMODEL;

    // fp32 buffers
    float* residual = ws;
    float* hidden   = residual + MD;
    float* xz       = hidden + MD;                          // [M,1536]
    float* xc       = xz    + (size_t)MROWS * 2 * DINNER;   // [M,768]
    float* dbl      = xc    + (size_t)MROWS * DINNER;       // [M,64] padded
    float* delta    = dbl   + (size_t)MROWS * DBLLD;        // [M,768]
    float* ybuf     = delta + (size_t)MROWS * DINNER;       // [M,768]
    float* fend     = ybuf  + (size_t)MROWS * DINNER;

    // bf16 staging buffers
    __bf16* hbf   = (__bf16*)fend;                          // [M,384]
    __bf16* xcbf  = hbf  + MD;                              // [M,768]
    __bf16* dblbf = xcbf + (size_t)MROWS * DINNER;          // [M,32] (24 + pad)
    __bf16* ybf   = dblbf + (size_t)MROWS * 32;             // [M,768]
    __bf16* wbf   = ybf  + (size_t)MROWS * DINNER;          // up to 1536x384

    build_kernel<<<(MROWS * DMODEL + 255) / 256, 256, 0, stream>>>(
        cjv, cls, pos, hidden, residual);

    for (int L = 0; L < NDEPTH; ++L) {
        rms_add_kernel<<<MROWS, 256, 0, stream>>>(residual, hidden,
                                                  nw + (size_t)L * DMODEL, hbf);

        // in_proj: [M,384] x [1536,384]^T -> xz [M,1536]
        launch_cvt(inW + (size_t)L * 2 * DINNER * DMODEL, DMODEL, wbf,
                   2 * DINNER, DMODEL, 2 * DINNER, DMODEL, stream);
        {
            dim3 g(MROWS / 16, (2 * DINNER) / 64);
            gemm_wmma_kernel<<<g, 32, 0, stream>>>(
                hbf, DMODEL, wbf, DMODEL, xz, 2 * DINNER, DMODEL, 0, nullptr);
        }

        for (int dir = 0; dir < 2; ++dir) {
            const size_t s = (size_t)(L * 2 + dir);
            conv_silu_kernel<<<(MROWS * DINNER + 255) / 256, 256, 0, stream>>>(
                xz, cW + s * DINNER * KCONV, cB + s * DINNER, xc, xcbf, dir);

            // x_proj: [M,768] x [56,768]^T -> dbl [M,64] (N padded 56->64)
            launch_cvt(xpW + s * 56 * DINNER, DINNER, wbf,
                       64, DINNER, 56, DINNER, stream);
            {
                dim3 g(MROWS / 16, 1);
                gemm_wmma_kernel<<<g, 32, 0, stream>>>(
                    xcbf, DINNER, wbf, DINNER, dbl, DBLLD, DINNER, 0, nullptr);
            }

            // stage dt input: dbl[:, :24] -> dblbf [M,32] (K padded 24->32)
            launch_cvt(dbl, DBLLD, dblbf, MROWS, 32, MROWS, DTRANK, stream);
            // dt_proj: [M,32] x [768,32]^T -> delta [M,768], softplus + bias
            launch_cvt(dtW + s * DINNER * DTRANK, DTRANK, wbf,
                       DINNER, 32, DINNER, DTRANK, stream);
            {
                dim3 g(MROWS / 16, DINNER / 64);
                gemm_wmma_kernel<<<g, 32, 0, stream>>>(
                    dblbf, 32, wbf, 32, delta, DINNER, 32, 1, dtB + s * DINNER);
            }

            scan_kernel<<<BATCH, DINNER, 0, stream>>>(
                xc, dbl, delta, xz, Alog + s * DINNER * DSTATE,
                Dsk + s * DINNER, ybuf, dir);
        }

        // out_proj: [M,768] x [384,768]^T -> hidden [M,384], *0.5
        launch_cvt(ybuf, DINNER, ybf, MROWS, DINNER, MROWS, DINNER, stream);
        launch_cvt(outW + (size_t)L * DMODEL * DINNER, DINNER, wbf,
                   DMODEL, DINNER, DMODEL, DINNER, stream);
        {
            dim3 g(MROWS / 16, DMODEL / 64);
            gemm_wmma_kernel<<<g, 32, 0, stream>>>(
                ybf, DINNER, wbf, DINNER, hidden, DMODEL, DINNER, 2, nullptr);
        }
    }

    final_kernel<<<MROWS, 256, 0, stream>>>(residual, hidden, fnw, out);
}
